// GateAttentionUnit_53042846106136
// MI455X (gfx1250) — matrix operand order repaired and checked by reading, not compiled, not run on hardware
//
#include <hip/hip_runtime.h>

// ---------------------------------------------------------------------------
// GAU (Gated Attention Unit) for MI455X / gfx1250, wave32 + WMMA bf16.
//   B=32, N=512, H=1024, E=2048, S=128, M=B*N=16384, UVN=2E+S=4224
// All GEMMs run through v_wmma_f32_16x16x32_bf16 (fp32 accumulate).
// Macro tile 256(M) x 64(N) x 32(K), 8 wave32: each wave owns a 32x64 slab
// (two 16-row A fragments sharing four B fragments -> 8 WMMA per K-step).
// LDS: A row-major, B column-major (per-lane frags contiguous -> ds_load_b128).
// Double-buffered LDS; staging kept in NAMED uint4 scalars (no arrays/lambdas)
// so the compiler keeps them in VGPRs instead of a scratch alloca.
// ---------------------------------------------------------------------------

typedef __attribute__((ext_vector_type(16))) __bf16 v16bf;
typedef __attribute__((ext_vector_type(8)))  float  v8f;

#define GB (32)
#define GN (512)
#define GH (1024)
#define GE (2048)
#define GS (128)
#define GM (GB * GN)          // 16384
#define UVN (2 * GE + GS)     // 4224

// ---------------------- small vector helpers -------------------------------
__device__ __forceinline__ uint4 ld16(const __bf16* p) {
    return *reinterpret_cast<const uint4*>(p);
}
__device__ __forceinline__ void st16(__bf16* p, uint4 v) {
    *reinterpret_cast<uint4*>(p) = v;
}
// Gather 8 bf16 down a column (stride ldb) into one 16-byte register.
__device__ __forceinline__ uint4 gather_bcol(const __bf16* bp, int ldb) {
    union { uint4 v; __bf16 h[8]; } u;
#pragma unroll
    for (int i = 0; i < 8; ++i) u.h[i] = bp[(long)i * ldb];
    return u.v;
}

// ---------------------- WMMA fragment helpers ------------------------------
// A (16x32 row-major, ld=32): lanes 0-15 -> row M=lane, K 0..7 / 16..23;
// lanes 16-31 -> row M=lane-16, K 8..15 / 24..31. Contiguous 8+8 -> 2x b128.
__device__ __forceinline__ v16bf frag_a_ld(const __bf16* a, int lane) {
    const int half = lane >> 4;
    const int row  = lane & 15;
    const __bf16* p = a + row * 32 + half * 8;
    v16bf f;
#pragma unroll
    for (int i = 0; i < 8; ++i) f[i] = p[i];
    p += 16;
#pragma unroll
    for (int i = 0; i < 8; ++i) f[8 + i] = p[i];
    return f;
}

// B stored column-major in LDS: bcm[n*32 + k] = B[k][n] for this K-chunk.
// Lane needs K = half*8 + 0..7 / 16 + half*8 + 0..7 at fixed N=col -> 2x b128.
__device__ __forceinline__ v16bf frag_b_cm(const __bf16* bcm, int lane) {
    const int half = lane >> 4;
    const int col  = lane & 15;
    const __bf16* p = bcm + col * 32 + half * 8;
    v16bf f;
#pragma unroll
    for (int i = 0; i < 8; ++i) f[i] = p[i];
    p += 16;
#pragma unroll
    for (int i = 0; i < 8; ++i) f[8 + i] = p[i];
    return f;
}

// ---------------------- Epilogue functors ----------------------------------
struct EpiUV {   // silu(acc + uv_b[n]) -> split into u (f32), v (bf16), base (f32)
    const float* bias; float* u; __bf16* v; float* base;
    __device__ void operator()(int /*bz*/, int m, int n, float val) const {
        val += bias[n];
        const float s = val / (1.0f + __expf(-val));
        if (n < GE)            u[(long)m * GE + n]             = s;
        else if (n < 2 * GE)   v[(long)m * GE + (n - GE)]      = (__bf16)s;
        else                   base[(long)m * GS + (n - 2*GE)] = s;
    }
};

struct EpiQK {   // relu(acc/512 + w[511+n-m] + (1-mask)*-1e12)^2 -> bf16
    const float* wrel; const float* mask; __bf16* ker;
    __device__ void operator()(int b, int m, int n, float val) const {
        float t = val * (1.0f / 512.0f) + wrel[511 + n - m]
                + (1.0f - mask[b * GN + n]) * -1.0e12f;
        t = fmaxf(t, 0.0f);
        ker[((long)b * GN + m) * GN + n] = (__bf16)(t * t);
    }
};

struct EpiAttn { // (kernel @ v) * u -> bf16 for the final GEMM
    const float* u; __bf16* out;
    __device__ void operator()(int b, int m, int n, float val) const {
        const long row = (long)b * GN + m;
        out[row * GE + n] = (__bf16)(val * u[row * GE + n]);
    }
};

struct EpiFinal { // acc + o_b[n] + shortcut x -> f32 output
    const float* ob; const float* x; float* y;
    __device__ void operator()(int /*bz*/, int m, int n, float val) const {
        y[(long)m * GH + n] = val + ob[n] + x[(long)m * GH + n];
    }
};

// ---------------------- Generic 256x64 WMMA GEMM ---------------------------
template <typename Epi>
__global__ __launch_bounds__(256) __attribute__((amdgpu_waves_per_eu(1)))
void gemm_wmma_256x64(
    const __bf16* __restrict__ A, const __bf16* __restrict__ B,
    int K, int lda, int ldb, long strideA, long strideB, Epi epi) {
    __shared__ __align__(16) __bf16 lA[2][256 * 32];   // row-major
    __shared__ __align__(16) __bf16 lB[2][64 * 32];    // column-major [n][k]

    const int t    = threadIdx.x;
    const int lane = t & 31;
    const int wave = t >> 5;

    const __bf16* Ab = A + (long)blockIdx.z * strideA + (long)blockIdx.x * 256 * lda;
    const __bf16* Bb = B + (long)blockIdx.z * strideB + (long)blockIdx.y * 64;

    // A tile copy: thread -> rows (t>>1) and (t>>1)+128, 16-col half (t&1).
    const int arow = t >> 1, acol = (t & 1) * 16;
    // B tile copy (transposing): thread -> column bcol, k rows brow..brow+7.
    // Global reads coalesce across lanes (consecutive n); LDS store is 1x b128.
    const int bcol = t & 63, brow = (t >> 6) * 8;

    const __bf16* apA = Ab + (long)arow * lda + acol;           // rows arow
    const __bf16* apB = apA + (long)128 * lda;                  // rows arow+128
    const __bf16* bp0 = Bb + (long)brow * ldb + bcol;

    __bf16* sA0 = &lA[0][arow * 32 + acol];
    __bf16* sA1 = &lA[0][(arow + 128) * 32 + acol];
    __bf16* sB  = &lB[0][bcol * 32 + brow];
    const int lAstep = 256 * 32;   // elements between buffers
    const int lBstep = 64 * 32;

    v8f acc0[4] = {};
    v8f acc1[4] = {};

    // ---- prologue: stage K-step 0 into buffer 0 ----
    uint4 ga0 = ld16(apA);
    uint4 ga1 = ld16(apA + 8);
    uint4 ga2 = ld16(apB);
    uint4 ga3 = ld16(apB + 8);
    uint4 gb  = gather_bcol(bp0, ldb);
    st16(sA0, ga0);
    st16(sA0 + 8, ga1);
    st16(sA1, ga2);
    st16(sA1 + 8, ga3);
    st16(sB, gb);
    __syncthreads();

    const int ksteps = K >> 5;
    for (int kt = 0; kt < ksteps; ++kt) {
        const int cur = kt & 1;
        const bool more = (kt + 1) < ksteps;
        if (more) {                         // issue next-tile loads early;
            const int k0 = (kt + 1) << 5;   // they complete during the WMMAs
            ga0 = ld16(apA + k0);
            ga1 = ld16(apA + k0 + 8);
            ga2 = ld16(apB + k0);
            ga3 = ld16(apB + k0 + 8);
            gb  = gather_bcol(bp0 + (long)k0 * ldb, ldb);
        }

        // Preload every fragment, then issue 8 back-to-back WMMAs so a single
        // dscnt wait covers all LDS traffic for this K-step.
        const __bf16* lab = &lA[0][cur * lAstep + (wave * 32) * 32];
        const __bf16* lbb = &lB[0][cur * lBstep];
        const v16bf af0 = frag_a_ld(lab, lane);
        const v16bf af1 = frag_a_ld(lab + 16 * 32, lane);
        v16bf bf0 = frag_b_cm(lbb, lane);
        v16bf bf1 = frag_b_cm(lbb + 16 * 32, lane);
        v16bf bf2 = frag_b_cm(lbb + 32 * 32, lane);
        v16bf bf3 = frag_b_cm(lbb + 48 * 32, lane);

        acc0[0] = __builtin_amdgcn_wmma_f32_16x16x32_bf16(false, af0, false, bf0, (short)0, acc0[0], false, false);
        acc0[1] = __builtin_amdgcn_wmma_f32_16x16x32_bf16(false, af0, false, bf1, (short)0, acc0[1], false, false);
        acc0[2] = __builtin_amdgcn_wmma_f32_16x16x32_bf16(false, af0, false, bf2, (short)0, acc0[2], false, false);
        acc0[3] = __builtin_amdgcn_wmma_f32_16x16x32_bf16(false, af0, false, bf3, (short)0, acc0[3], false, false);
        acc1[0] = __builtin_amdgcn_wmma_f32_16x16x32_bf16(false, af1, false, bf0, (short)0, acc1[0], false, false);
        acc1[1] = __builtin_amdgcn_wmma_f32_16x16x32_bf16(false, af1, false, bf1, (short)0, acc1[1], false, false);
        acc1[2] = __builtin_amdgcn_wmma_f32_16x16x32_bf16(false, af1, false, bf2, (short)0, acc1[2], false, false);
        acc1[3] = __builtin_amdgcn_wmma_f32_16x16x32_bf16(false, af1, false, bf3, (short)0, acc1[3], false, false);

        if (more) {
            __syncthreads();              // everyone done reading buf cur^1
            const int nxt = (cur ^ 1);
            st16(sA0 + nxt * lAstep, ga0);
            st16(sA0 + nxt * lAstep + 8, ga1);
            st16(sA1 + nxt * lAstep, ga2);
            st16(sA1 + nxt * lAstep + 8, ga3);
            st16(sB + nxt * lBstep, gb);
            __syncthreads();              // tiles visible before next compute
        }
    }

    // C/D layout: VGPR r -> row (half*8 + r), col = lane&15 within subtile.
    const int half = lane >> 4, col = lane & 15;
    const int gm0  = blockIdx.x * 256 + wave * 32 + half * 8;
    const int gn0  = blockIdx.y * 64;
#pragma unroll
    for (int j = 0; j < 4; ++j)
#pragma unroll
        for (int r = 0; r < 8; ++r) {
            epi((int)blockIdx.z, gm0 + r,      gn0 + j * 16 + col, acc0[j][r]);
            epi((int)blockIdx.z, gm0 + 16 + r, gn0 + j * 16 + col, acc1[j][r]);
        }
}

// ---------------------- LayerNorm (fused f32 -> bf16) ----------------------
__global__ __launch_bounds__(256) void layernorm_bf16(
    const float* __restrict__ x, const float* __restrict__ g,
    const float* __restrict__ b, __bf16* __restrict__ xn) {
    const long row = blockIdx.x;
    const int  t   = threadIdx.x;
    const float4 xv = reinterpret_cast<const float4*>(x + row * GH)[t];
    float s  = xv.x + xv.y + xv.z + xv.w;
    float s2 = xv.x * xv.x + xv.y * xv.y + xv.z * xv.z + xv.w * xv.w;

    __shared__ float rs[256], rq[256];
    rs[t] = s; rq[t] = s2;
    __syncthreads();
    for (int o = 128; o > 0; o >>= 1) {
        if (t < o) { rs[t] += rs[t + o]; rq[t] += rq[t + o]; }
        __syncthreads();
    }
    const float mu  = rs[0] * (1.0f / GH);
    const float var = rq[0] * (1.0f / GH) - mu * mu;
    const float inv = rsqrtf(var + 1e-5f);

    const float4 gv = reinterpret_cast<const float4*>(g)[t];
    const float4 bv = reinterpret_cast<const float4*>(b)[t];
    __bf16* o = xn + row * GH + t * 4;
    o[0] = (__bf16)((xv.x - mu) * inv * gv.x + bv.x);
    o[1] = (__bf16)((xv.y - mu) * inv * gv.y + bv.y);
    o[2] = (__bf16)((xv.z - mu) * inv * gv.z + bv.z);
    o[3] = (__bf16)((xv.w - mu) * inv * gv.w + bv.w);
}

// ---------------------- RoPE + affine -> q (bf16), kT (bf16) ---------------
// grid (N, B), 64 threads (d = 0..63 of the 64-wide half).
__global__ void qk_prep(const float* __restrict__ base,
                        const float* __restrict__ gamma,
                        const float* __restrict__ beta,
                        __bf16* __restrict__ q, __bf16* __restrict__ kT) {
    const int n = blockIdx.x, b = blockIdx.y, d = threadIdx.x;
    const float* br = base + ((long)b * GN + n) * GS;
    const float inv = __powf(10000.0f, (float)d * (1.0f / 64.0f));
    const float ang = (float)n * inv;
    const float sn = __sinf(ang), cs = __cosf(ang);

    // q: gamma/beta row 0
    const float qx1 = br[d]      * gamma[d]      + beta[d];
    const float qx2 = br[d + 64] * gamma[d + 64] + beta[d + 64];
    __bf16* qr = q + ((long)b * GN + n) * GS;
    qr[d]      = (__bf16)(qx1 * cs - qx2 * sn);
    qr[d + 64] = (__bf16)(qx2 * cs + qx1 * sn);

    // k: gamma/beta row 1; store transposed (B, 128, 512) for the qk GEMM
    const float kx1 = br[d]      * gamma[GS + d]      + beta[GS + d];
    const float kx2 = br[d + 64] * gamma[GS + d + 64] + beta[GS + d + 64];
    __bf16* kb = kT + (long)b * GS * GN;
    kb[(long)d * GN + n]        = (__bf16)(kx1 * cs - kx2 * sn);
    kb[(long)(d + 64) * GN + n] = (__bf16)(kx2 * cs + kx1 * sn);
}

// ---------------------- f32 -> bf16 elementwise ----------------------------
__global__ void f32_to_bf16(const float* __restrict__ s, __bf16* __restrict__ d, int n) {
    const int i = blockIdx.x * blockDim.x + threadIdx.x;
    if (i < n) d[i] = (__bf16)s[i];
}

// ---------------------- Host orchestration ---------------------------------
extern "C" void kernel_launch(void* const* d_in, const int* /*in_sizes*/, int /*n_in*/,
                              void* d_out, int /*out_size*/, void* d_ws, size_t /*ws_size*/,
                              hipStream_t stream) {
    const float* x    = (const float*)d_in[0];
    const float* mask = (const float*)d_in[1];
    const float* gamma= (const float*)d_in[2];
    const float* beta = (const float*)d_in[3];
    const float* w    = (const float*)d_in[4];
    // d_in[5], d_in[6] (a_p, b_p) unused by the reference
    const float* uvw  = (const float*)d_in[7];
    const float* uvb  = (const float*)d_in[8];
    const float* ow   = (const float*)d_in[9];
    const float* ob   = (const float*)d_in[10];
    const float* lng  = (const float*)d_in[11];
    const float* lnb  = (const float*)d_in[12];
    float* y = (float*)d_out;

    char* ws = (char*)d_ws;
    size_t off = 0;
    auto take = [&](size_t bytes) -> void* {
        void* p = ws + off;
        off = (off + bytes + 255) & ~(size_t)255;
        return p;
    };
    __bf16* xn_b   = (__bf16*)take((size_t)GM * GH * 2);        //  32 MB
    __bf16* uvw_b  = (__bf16*)take((size_t)GH * UVN * 2);       // 8.3 MB
    __bf16* ow_b   = (__bf16*)take((size_t)GE * GH * 2);        //   4 MB
    float*  u_f    = (float*) take((size_t)GM * GE * 4);        // 128 MB
    __bf16* v_b    = (__bf16*)take((size_t)GM * GE * 2);        //  64 MB
    float*  base_f = (float*) take((size_t)GM * GS * 4);        //   8 MB
    __bf16* q_b    = (__bf16*)take((size_t)GM * GS * 2);        //   4 MB
    __bf16* kT_b   = (__bf16*)take((size_t)GB * GS * GN * 2);   //   4 MB
    __bf16* ker_b  = (__bf16*)take((size_t)GB * GN * GN * 2);   //  16 MB
    __bf16* attn_b = (__bf16*)take((size_t)GM * GE * 2);        //  64 MB

    // 1) Weights -> bf16
    f32_to_bf16<<<(GH * UVN + 255) / 256, 256, 0, stream>>>(uvw, uvw_b, GH * UVN);
    f32_to_bf16<<<(GE * GH + 255) / 256, 256, 0, stream>>>(ow, ow_b, GE * GH);

    // 2) LayerNorm (fused bf16 convert)
    layernorm_bf16<<<GM, 256, 0, stream>>>(x, lng, lnb, xn_b);

    // 3) uv = silu(xn @ uv_w + uv_b); split -> u (f32), v (bf16), base (f32)
    gemm_wmma_256x64<<<dim3(GM / 256, UVN / 64, 1), 256, 0, stream>>>(
        xn_b, uvw_b, GH, GH, UVN, 0L, 0L, EpiUV{uvb, u_f, v_b, base_f});

    // 4) RoPE + per-head affine -> q, kT
    qk_prep<<<dim3(GN, GB), 64, 0, stream>>>(base_f, gamma, beta, q_b, kT_b);

    // 5) kernel = relu(q kT / 512 + relbias + maskpen)^2  (batched)
    gemm_wmma_256x64<<<dim3(GN / 256, GN / 64, GB), 256, 0, stream>>>(
        q_b, kT_b, GS, GS, GN, (long)GN * GS, (long)GS * GN, EpiQK{w, mask, ker_b});

    // 6) attn = u * (kernel @ v)  (batched)
    gemm_wmma_256x64<<<dim3(GN / 256, GE / 64, GB), 256, 0, stream>>>(
        ker_b, v_b, GN, GN, GE, (long)GN * GN, (long)GN * GE, EpiAttn{u_f, attn_b});

    // 7) y = attn @ o_w + o_b + x
    gemm_wmma_256x64<<<dim3(GM / 256, GH / 64, 1), 256, 0, stream>>>(
        attn_b, ow_b, GE, GE, GH, 0L, 0L, EpiFinal{ob, x, y});
}
